// Block_87351044866235
// MI455X (gfx1250) — matrix operand order repaired and checked by reading, not compiled, not run on hardware
//
#include <hip/hip_runtime.h>
#include <math.h>

typedef __attribute__((ext_vector_type(16))) _Float16 v16h;
typedef __attribute__((ext_vector_type(8)))  float    v8f;

#define B_   8
#define H_   128
#define W_   128
#define NSP  16384      // H*W
#define TT   131072     // B*NSP
#define CE   192
#define CP   96
#define CQ   288
#define NH   8
#define CPH  12
#define HIDD 256

static __device__ __forceinline__ float gelu_exact(float x) {
  return 0.5f * x * (1.0f + erff(x * 0.7071067811865475f));
}

// ---------------- depthwise 3x3 + residual (pos embed), NHWC C=192 ----------
__global__ void pos_dw_kernel(const float* __restrict__ x,
                              const float* __restrict__ w,
                              const float* __restrict__ bias,
                              float* __restrict__ out) {
  long idx = (long)blockIdx.x * blockDim.x + threadIdx.x;
  if (idx >= (long)TT * CE) return;
  int c = (int)(idx % CE);
  long t = idx / CE;
  int b = (int)(t >> 14);
  int n = (int)(t & (NSP - 1));
  int h = n >> 7, ww = n & 127;
  const float* wk = w + c * 9;
  float s = bias[c];
  #pragma unroll
  for (int kh = 0; kh < 3; ++kh) {
    int hy = h + kh - 1;
    if (hy < 0 || hy >= H_) continue;
    #pragma unroll
    for (int kw = 0; kw < 3; ++kw) {
      int wx = ww + kw - 1;
      if (wx < 0 || wx >= W_) continue;
      s += x[(((long)b * NSP + hy * W_ + wx) * CE) + c] * wk[kh * 3 + kw];
    }
  }
  out[idx] = x[idx] + s;
}

// ---------------- depthwise 3x3 (qkv dw), NHWC C=288, no bias ---------------
__global__ void qkv_dw_kernel(const float* __restrict__ x,
                              const float* __restrict__ w,
                              float* __restrict__ out) {
  long idx = (long)blockIdx.x * blockDim.x + threadIdx.x;
  if (idx >= (long)TT * CQ) return;
  int c = (int)(idx % CQ);
  long t = idx / CQ;
  int b = (int)(t >> 14);
  int n = (int)(t & (NSP - 1));
  int h = n >> 7, ww = n & 127;
  const float* wk = w + c * 9;
  float s = 0.f;
  #pragma unroll
  for (int kh = 0; kh < 3; ++kh) {
    int hy = h + kh - 1;
    if (hy < 0 || hy >= H_) continue;
    #pragma unroll
    for (int kw = 0; kw < 3; ++kw) {
      int wx = ww + kw - 1;
      if (wx < 0 || wx >= W_) continue;
      s += x[(((long)b * NSP + hy * W_ + wx) * CQ) + c] * wk[kh * 3 + kw];
    }
  }
  out[idx] = s;
}

// ---------------- LayerNorm over 192 channels, one wave per token ----------
__global__ void ln_kernel(const float* __restrict__ in,
                          const float* __restrict__ g,
                          const float* __restrict__ bta,
                          float* __restrict__ out) {
  int wave = (int)((blockIdx.x * (long)blockDim.x + threadIdx.x) >> 5);
  int lane = threadIdx.x & 31;
  if (wave >= TT) return;
  const float* row = in + (long)wave * CE;
  float v[6], s = 0.f;
  #pragma unroll
  for (int i = 0; i < 6; ++i) { v[i] = row[lane + 32 * i]; s += v[i]; }
  #pragma unroll
  for (int o = 16; o; o >>= 1) s += __shfl_xor(s, o, 32);
  float m = s * (1.0f / CE);
  float q = 0.f;
  #pragma unroll
  for (int i = 0; i < 6; ++i) { float d = v[i] - m; q += d * d; }
  #pragma unroll
  for (int o = 16; o; o >>= 1) q += __shfl_xor(q, o, 32);
  float inv = rsqrtf(q * (1.0f / CE) + 1e-6f);
  float* orow = out + (long)wave * CE;
  #pragma unroll
  for (int i = 0; i < 6; ++i) {
    int c = lane + 32 * i;
    orow[c] = (v[i] - m) * inv * g[c] + bta[c];
  }
}

// -------- WMMA fragment loaders (f32 -> f16), b128 global loads ------------
// A 16x32 f16 layout: lane l: M = l&15, g = l>>4; af[2r+j] <-> K = (r<4?0:16)+8g+2(r&3)+j
static __device__ __forceinline__ v16h load_a_frag(const float* __restrict__ arow,
                                                   int k0, int g) {
  const float4* p0 = (const float4*)(arow + k0 + 8 * g);
  const float4* p2 = (const float4*)(arow + k0 + 16 + 8 * g);
  float4 f0 = p0[0], f1 = p0[1], f2 = p2[0], f3 = p2[1];
  v16h a;
  a[0]  = (_Float16)f0.x; a[1]  = (_Float16)f0.y;
  a[2]  = (_Float16)f0.z; a[3]  = (_Float16)f0.w;
  a[4]  = (_Float16)f1.x; a[5]  = (_Float16)f1.y;
  a[6]  = (_Float16)f1.z; a[7]  = (_Float16)f1.w;
  a[8]  = (_Float16)f2.x; a[9]  = (_Float16)f2.y;
  a[10] = (_Float16)f2.z; a[11] = (_Float16)f2.w;
  a[12] = (_Float16)f3.x; a[13] = (_Float16)f3.y;
  a[14] = (_Float16)f3.z; a[15] = (_Float16)f3.w;
  return a;
}
// B 32x16 f16 layout: lane l: N = l&15, g = l>>4; bf[2r+j] <-> K = (r<4?0:16)+8g+2(r&3)+j
static __device__ __forceinline__ v16h load_b_frag(const float* __restrict__ brow,
                                                   int k0, int g) {
  const float4* p = (const float4*)(brow + k0 + 8 * g);
  const float4* q = (const float4*)(brow + k0 + 16 + 8 * g);
  float4 f0 = p[0], f1 = p[1], f2 = q[0], f3 = q[1];
  v16h b;
  b[0]  = (_Float16)f0.x; b[1]  = (_Float16)f0.y;
  b[2]  = (_Float16)f0.z; b[3]  = (_Float16)f0.w;
  b[4]  = (_Float16)f1.x; b[5]  = (_Float16)f1.y;
  b[6]  = (_Float16)f1.z; b[7]  = (_Float16)f1.w;
  b[8]  = (_Float16)f2.x; b[9]  = (_Float16)f2.y;
  b[10] = (_Float16)f2.z; b[11] = (_Float16)f2.w;
  b[12] = (_Float16)f3.x; b[13] = (_Float16)f3.y;
  b[14] = (_Float16)f3.z; b[15] = (_Float16)f3.w;
  return b;
}

// ---------------- WMMA f16 GEMM, 32x32 per wave (2x2 tiles) ----------------
// out[M,N] = act[M,K(lda)] @ wt[N,K]^T
// EPI: 0 plain, 1 +resid (in-place ok), 2 gelu+BN, 3 BN+resid
template <int EPI>
__global__ void gemm_wmma(const float* __restrict__ act, int lda,
                          const float* __restrict__ wt,
                          float* __restrict__ out, int ldo,
                          const float* __restrict__ resid,
                          const float* __restrict__ bng,
                          const float* __restrict__ bnb,
                          const float* __restrict__ bnm,
                          const float* __restrict__ bnv,
                          int K, int ntn2, int ntiles) {
  int wv = (int)((blockIdx.x * (long)blockDim.x + threadIdx.x) >> 5);
  if (wv >= ntiles) return;               // wave-uniform: EXEC stays all-ones
  int lane = threadIdx.x & 31;
  int tn2 = wv % ntn2, tm2 = wv / ntn2;
  int l15 = lane & 15, g = lane >> 4;
  const float* arow0 = act + (long)(tm2 * 32 + l15) * lda;
  const float* arow1 = arow0 + (long)16 * lda;
  const float* brow0 = wt + (long)(tn2 * 32 + l15) * K;
  const float* brow1 = brow0 + (long)16 * K;
  v8f acc[2][2] = {};
  for (int k0 = 0; k0 < K; k0 += 32) {
    v16h a0 = load_a_frag(arow0, k0, g);
    v16h a1 = load_a_frag(arow1, k0, g);
    v16h b0 = load_b_frag(brow0, k0, g);
    v16h b1 = load_b_frag(brow1, k0, g);
    acc[0][0] = __builtin_amdgcn_wmma_f32_16x16x32_f16(false, a0, false, b0,
                                                       (short)0, acc[0][0], false, false);
    acc[0][1] = __builtin_amdgcn_wmma_f32_16x16x32_f16(false, a0, false, b1,
                                                       (short)0, acc[0][1], false, false);
    acc[1][0] = __builtin_amdgcn_wmma_f32_16x16x32_f16(false, a1, false, b0,
                                                       (short)0, acc[1][0], false, false);
    acc[1][1] = __builtin_amdgcn_wmma_f32_16x16x32_f16(false, a1, false, b1,
                                                       (short)0, acc[1][1], false, false);
  }
  #pragma unroll
  for (int j = 0; j < 2; ++j) {
    int nn = tn2 * 32 + 16 * j + l15;     // D layout: N = lane & 15
    float bsc = 0.f, bmn = 0.f, bbt = 0.f;
    if constexpr (EPI == 2 || EPI == 3) {
      bsc = bng[nn] * rsqrtf(bnv[nn] + 1e-5f);
      bmn = bnm[nn];
      bbt = bnb[nn];
    }
    #pragma unroll
    for (int i = 0; i < 2; ++i) {
      #pragma unroll
      for (int r = 0; r < 8; ++r) {
        int mm = tm2 * 32 + 16 * i + r + 8 * g;   // D layout: M = r + 8*g
        long o = (long)mm * ldo + nn;
        float vv = acc[i][j][r];
        if constexpr (EPI == 1) {
          vv += resid[o];
        } else if constexpr (EPI == 2) {
          vv = gelu_exact(vv);
          vv = (vv - bmn) * bsc + bbt;
        } else if constexpr (EPI == 3) {
          vv = (vv - bmn) * bsc + bbt;
          vv += resid[o];
        }
        out[o] = vv;
      }
    }
  }
}

// ---------------- q/k L2 norm scales: one block per (b, c<192) --------------
__global__ void qknorm_kernel(const float* __restrict__ q1,
                              float* __restrict__ scale) {
  int b = blockIdx.x / CE, c = blockIdx.x % CE;
  const float* base = q1 + (long)b * NSP * CQ + c;
  float s = 0.f;
  for (int n = threadIdx.x; n < NSP; n += blockDim.x) {
    float v = base[(long)n * CQ];
    s += v * v;
  }
  __shared__ float sm[8];
  int lane = threadIdx.x & 31, w = threadIdx.x >> 5;
  #pragma unroll
  for (int o = 16; o; o >>= 1) s += __shfl_xor(s, o, 32);
  if (lane == 0) sm[w] = s;
  __syncthreads();
  if (threadIdx.x == 0) {
    float tot = 0.f;
    for (int i = 0; i < 8; ++i) tot += sm[i];
    scale[blockIdx.x] = 1.0f / fmaxf(sqrtf(tot), 1e-12f);
  }
}

// ---------------- attn[b,hd,c,d]: one block per output element -------------
__global__ void attn_kernel(const float* __restrict__ q1,
                            const float* __restrict__ scale,
                            const float* __restrict__ temp,
                            float* __restrict__ attn) {
  int idx = blockIdx.x;               // 8*8*12*12 = 9216
  int d  = idx % CPH;
  int c  = (idx / CPH) % CPH;
  int hd = (idx / (CPH * CPH)) % NH;
  int b  = idx / (CPH * CPH * NH);
  int qc = hd * CPH + c;
  int kc = CP + hd * CPH + d;
  const float* base = q1 + (long)b * NSP * CQ;
  float s = 0.f;
  for (int n = threadIdx.x; n < NSP; n += blockDim.x)
    s += base[(long)n * CQ + qc] * base[(long)n * CQ + kc];
  __shared__ float sm[8];
  int lane = threadIdx.x & 31, w = threadIdx.x >> 5;
  #pragma unroll
  for (int o = 16; o; o >>= 1) s += __shfl_xor(s, o, 32);
  if (lane == 0) sm[w] = s;
  __syncthreads();
  if (threadIdx.x == 0) {
    float tot = 0.f;
    for (int i = 0; i < 8; ++i) tot += sm[i];
    attn[idx] = tot * scale[b * CE + qc] * scale[b * CE + kc] * temp[hd];
  }
}

// ---------------- gate: per-token MLP + sigmoid, global mean accumulation --
__global__ void gate_kernel(const float* __restrict__ y,
                            const float* __restrict__ w1,
                            const float* __restrict__ b1,
                            const float* __restrict__ w2,
                            const float* __restrict__ b2,
                            float* __restrict__ accum) {
  int t = blockIdx.x * blockDim.x + threadIdx.x;
  float sg = 0.f;
  if (t < TT) {
    const float* row = y + (long)t * CE;
    float g2 = b2[0];
    for (int j = 0; j < CP; ++j) {
      const float* wr = w1 + j * CE;
      float a = b1[j];
      for (int cc = 0; cc < CE; ++cc) a += row[cc] * wr[cc];
      g2 += fmaxf(a, 0.f) * w2[j];
    }
    sg = 1.0f / (1.0f + expf(-g2));
  }
  #pragma unroll
  for (int o = 16; o; o >>= 1) sg += __shfl_xor(sg, o, 32);
  if ((threadIdx.x & 31) == 0) atomicAdd(accum, sg);
}

// ---------------- dynamic top-k mask + softmax over 12-entry rows ----------
__global__ void topk_softmax_kernel(float* __restrict__ attn,
                                    const float* __restrict__ accum) {
  int row = blockIdx.x * blockDim.x + threadIdx.x;
  if (row >= B_ * NH * CPH) return;     // 768 rows
  float dynk = floorf((float)CPH * (accum[0] / (float)TT));
  float* a = attn + row * CPH;
  float v[CPH], msk[CPH];
  #pragma unroll
  for (int i = 0; i < CPH; ++i) v[i] = a[i];
  float mx = -INFINITY;
  #pragma unroll
  for (int i = 0; i < CPH; ++i) {
    int rank = 0;
    for (int j = 0; j < CPH; ++j)
      rank += (v[j] > v[i]) || (v[j] == v[i] && j < i);   // stable argsort tie-break
    msk[i] = ((float)rank < dynk) ? v[i] : -INFINITY;
    mx = fmaxf(mx, msk[i]);
  }
  float sum = 0.f, e[CPH];
  #pragma unroll
  for (int i = 0; i < CPH; ++i) { e[i] = expf(msk[i] - mx); sum += e[i]; }
  #pragma unroll
  for (int i = 0; i < CPH; ++i) a[i] = e[i] / sum;
}

// ---------------- attn @ v, scale, write to Y[:, :96] ----------------------
__global__ void attnout_kernel(const float* __restrict__ q1,
                               const float* __restrict__ attn,
                               const float* __restrict__ a1,
                               const float* __restrict__ a2,
                               const float* __restrict__ a3,
                               const float* __restrict__ a4,
                               float* __restrict__ y) {
  long idx = (long)blockIdx.x * blockDim.x + threadIdx.x;
  if (idx >= (long)TT * CP) return;
  int c96 = (int)(idx % CP);
  long t = idx / CP;
  int b = (int)(t >> 14);
  int hd = c96 / CPH, c = c96 % CPH;
  const float* ar = attn + (((b * NH + hd) * CPH) + c) * CPH;
  const float* vb = q1 + (long)t * CQ + 2 * CP + hd * CPH;   // v channels
  float s = 0.f;
  #pragma unroll
  for (int d = 0; d < CPH; ++d) s += ar[d] * vb[d];
  s *= (a1[0] + a2[0] + a3[0] + a4[0]);
  y[t * CE + c96] = s;
}

// ---------------- multi-scale depthwise (k=1,3,5,7) + v0 residual ----------
__global__ void msdw_kernel(const float* __restrict__ v0,
                            const float* __restrict__ w0, const float* __restrict__ bb0,
                            const float* __restrict__ w1, const float* __restrict__ bb1,
                            const float* __restrict__ w2, const float* __restrict__ bb2,
                            const float* __restrict__ w3, const float* __restrict__ bb3,
                            float* __restrict__ u) {
  long idx = (long)blockIdx.x * blockDim.x + threadIdx.x;
  if (idx >= (long)TT * HIDD) return;
  int c = (int)(idx % HIDD);
  long t = idx / HIDD;
  int b = (int)(t >> 14);
  int n = (int)(t & (NSP - 1));
  int h = n >> 7, ww = n & 127;
  int grp = c >> 6, lc = c & 63;
  const float* wp; const float* bp; int S;
  if (grp == 0)      { wp = w0; bp = bb0; S = 1; }
  else if (grp == 1) { wp = w1; bp = bb1; S = 3; }
  else if (grp == 2) { wp = w2; bp = bb2; S = 5; }
  else               { wp = w3; bp = bb3; S = 7; }
  int pad = S >> 1;
  const float* wk = wp + lc * S * S;
  float s = bp[lc];
  for (int kh = 0; kh < S; ++kh) {
    int hy = h + kh - pad;
    if (hy < 0 || hy >= H_) continue;
    for (int kw = 0; kw < S; ++kw) {
      int wx = ww + kw - pad;
      if (wx < 0 || wx >= W_) continue;
      s += v0[(((long)b * NSP + hy * W_ + wx) * HIDD) + c] * wk[kh * S + kw];
    }
  }
  u[idx] = s + v0[idx];
}

// ---------------- gelu + BN2 then multiply by v0 (in-place on u) -----------
__global__ void gbm_kernel(float* __restrict__ u,
                           const float* __restrict__ v0,
                           const float* __restrict__ g,
                           const float* __restrict__ bb,
                           const float* __restrict__ m,
                           const float* __restrict__ vv) {
  long idx = (long)blockIdx.x * blockDim.x + threadIdx.x;
  if (idx >= (long)TT * HIDD) return;
  int c = (int)(idx % HIDD);
  float x = gelu_exact(u[idx]);
  x = (x - m[c]) * (g[c] * rsqrtf(vv[c] + 1e-5f)) + bb[c];
  u[idx] = x * v0[idx];
}

__global__ void zero_kernel(float* __restrict__ p, int n) {
  int i = blockIdx.x * blockDim.x + threadIdx.x;
  if (i < n) p[i] = 0.f;
}

extern "C" void kernel_launch(void* const* d_in, const int* in_sizes, int n_in,
                              void* d_out, int out_size, void* d_ws, size_t ws_size,
                              hipStream_t stream) {
  const float* x      = (const float*)d_in[0];
  const float* pos_w  = (const float*)d_in[1];
  const float* pos_b  = (const float*)d_in[2];
  const float* ln1_g  = (const float*)d_in[3];
  const float* ln1_b  = (const float*)d_in[4];
  const float* temp   = (const float*)d_in[5];
  const float* qkv_w  = (const float*)d_in[6];
  const float* qkv_dw = (const float*)d_in[7];
  const float* proj_w = (const float*)d_in[8];
  const float* gw1    = (const float*)d_in[9];
  const float* gb1    = (const float*)d_in[10];
  const float* gw2    = (const float*)d_in[11];
  const float* gb2    = (const float*)d_in[12];
  const float* at1    = (const float*)d_in[13];
  const float* at2    = (const float*)d_in[14];
  const float* at3    = (const float*)d_in[15];
  const float* at4    = (const float*)d_in[16];
  const float* ln2_g  = (const float*)d_in[17];
  const float* ln2_b  = (const float*)d_in[18];
  const float* fc1_w  = (const float*)d_in[19];
  const float* bn1g   = (const float*)d_in[20];
  const float* bn1b   = (const float*)d_in[21];
  const float* bn1m   = (const float*)d_in[22];
  const float* bn1v   = (const float*)d_in[23];
  const float* bn2g   = (const float*)d_in[24];
  const float* bn2b   = (const float*)d_in[25];
  const float* bn2m   = (const float*)d_in[26];
  const float* bn2v   = (const float*)d_in[27];
  const float* fc2_w  = (const float*)d_in[28];
  const float* bn3g   = (const float*)d_in[29];
  const float* bn3b   = (const float*)d_in[30];
  const float* bn3m   = (const float*)d_in[31];
  const float* bn3v   = (const float*)d_in[32];
  const float* dw0w   = (const float*)d_in[33];
  const float* dw0b   = (const float*)d_in[34];
  const float* dw1w   = (const float*)d_in[35];
  const float* dw1b   = (const float*)d_in[36];
  const float* dw2w   = (const float*)d_in[37];
  const float* dw2b   = (const float*)d_in[38];
  const float* dw3w   = (const float*)d_in[39];
  const float* dw3b   = (const float*)d_in[40];

  float* out = (float*)d_out;
  float* ws  = (float*)d_ws;
  float* A  = ws;                      // xc (residual stream), T*192
  float* Y  = A  + (long)TT * CE;      // LN out / [attn_out | x2], T*192
  float* Q0 = Y  + (long)TT * CE;      // qkv pre-dw, later u, T*288
  float* Q1 = Q0 + (long)TT * CQ;      // qkv post-dw, later v0, T*288
  float* SC = Q1 + (long)TT * CQ;      // q/k inverse norms, 8*192
  float* AT = SC + B_ * CE;            // attention matrix, 9216
  float* AC = AT + B_ * NH * CPH * CPH; // gate-mean accumulator, 1

  const int TPB = 256;

  // 1. positional depthwise conv + residual  -> A
  pos_dw_kernel<<<(int)(((long)TT * CE + TPB - 1) / TPB), TPB, 0, stream>>>(
      x, pos_w, pos_b, A);
  // 2. LN1  A -> Y
  ln_kernel<<<TT * 32 / TPB, TPB, 0, stream>>>(A, ln1_g, ln1_b, Y);
  // 3. gate mean (needs full Y before columns 0..95 are overwritten)
  zero_kernel<<<1, 32, 0, stream>>>(AC, 1);
  gate_kernel<<<TT / TPB, TPB, 0, stream>>>(Y, gw1, gb1, gw2, gb2, AC);
  // 4. qkv 1x1 GEMM (WMMA): Y[:, :96] @ qkv_w^T -> Q0 [T,288]
  {
    int ntn2 = CQ / 32, ntiles = (TT / 32) * ntn2;
    gemm_wmma<0><<<(ntiles * 32 + TPB - 1) / TPB, TPB, 0, stream>>>(
        Y, CE, qkv_w, Q0, CQ, nullptr, nullptr, nullptr, nullptr, nullptr,
        CP, ntn2, ntiles);
  }
  // 5. qkv depthwise 3x3: Q0 -> Q1
  qkv_dw_kernel<<<(int)(((long)TT * CQ + TPB - 1) / TPB), TPB, 0, stream>>>(
      Q0, qkv_dw, Q1);
  // 6. q/k L2 norms -> SC
  qknorm_kernel<<<B_ * CE, TPB, 0, stream>>>(Q1, SC);
  // 7. attn = norm(q) @ norm(k)^T * temperature -> AT
  attn_kernel<<<B_ * NH * CPH * CPH, TPB, 0, stream>>>(Q1, SC, temp, AT);
  // 8. dynamic top-k + softmax (in place on AT)
  topk_softmax_kernel<<<3, TPB, 0, stream>>>(AT, AC);
  // 9. attn @ v, scale, write into Y[:, :96] (x2 stays in Y[:, 96:192])
  attnout_kernel<<<(int)(((long)TT * CP + TPB - 1) / TPB), TPB, 0, stream>>>(
      Q1, AT, at1, at2, at3, at4, Y);
  // 10. proj GEMM (WMMA) + residual: A += Y @ proj_w^T
  {
    int ntn2 = CE / 32, ntiles = (TT / 32) * ntn2;
    gemm_wmma<1><<<(ntiles * 32 + TPB - 1) / TPB, TPB, 0, stream>>>(
        Y, CE, proj_w, A, CE, A, nullptr, nullptr, nullptr, nullptr,
        CE, ntn2, ntiles);
  }
  // 11. LN2  A -> Y
  ln_kernel<<<TT * 32 / TPB, TPB, 0, stream>>>(A, ln2_g, ln2_b, Y);
  // 12. fc1 GEMM (WMMA) + gelu + BN1 -> v0 in Q1 [T,256]
  {
    int ntn2 = HIDD / 32, ntiles = (TT / 32) * ntn2;
    gemm_wmma<2><<<(ntiles * 32 + TPB - 1) / TPB, TPB, 0, stream>>>(
        Y, CE, fc1_w, Q1, HIDD, nullptr, bn1g, bn1b, bn1m, bn1v,
        CE, ntn2, ntiles);
  }
  // 13. multi-scale depthwise + v0 residual: Q1 -> Q0 [T,256]
  msdw_kernel<<<(int)(((long)TT * HIDD + TPB - 1) / TPB), TPB, 0, stream>>>(
      Q1, dw0w, dw0b, dw1w, dw1b, dw2w, dw2b, dw3w, dw3b, Q0);
  // 14. gelu + BN2, multiply by v0 (in place on Q0)
  gbm_kernel<<<(int)(((long)TT * HIDD + TPB - 1) / TPB), TPB, 0, stream>>>(
      Q0, Q1, bn2g, bn2b, bn2m, bn2v);
  // 15. fc2 GEMM (WMMA) + BN3 + residual(A) -> d_out [T,192] (NHWC)
  {
    int ntn2 = CE / 32, ntiles = (TT / 32) * ntn2;
    gemm_wmma<3><<<(ntiles * 32 + TPB - 1) / TPB, TPB, 0, stream>>>(
        Q0, HIDD, fc2_w, out, CE, A, bn3g, bn3b, bn3m, bn3v,
        HIDD, ntn2, ntiles);
  }
}
